// Net_82068235092725
// MI455X (gfx1250) — matrix operand order repaired and checked by reading, not compiled, not run on hardware
//
#include <hip/hip_runtime.h>

typedef __attribute__((ext_vector_type(16))) _Float16 v16h;
typedef __attribute__((ext_vector_type(8)))  _Float16 v8h;
typedef __attribute__((ext_vector_type(8)))  float    v8f;
typedef __attribute__((ext_vector_type(4)))  float    v4f;

#define KDIM 128   // all GEMMs in this net have K = 128

// ---------------------------------------------------------------------------
// A-fragment: 16x32 f16 tile of row-major f32 matrix, converted on the fly.
// Lane l (wave32): M = l&15 ; K-halves: elems 0..7 = K kb..kb+7,
// elems 8..15 = K kb+16..kb+23, where kb = kt*32 + (l>=16 ? 8 : 0).
// 'scale' lets us fuse the scatter-mean divide into the operand load.
// ---------------------------------------------------------------------------
__device__ inline v16h load_a_frag(const float* __restrict__ row, int kt,
                                   int lane, float scale) {
  int kb = kt * 32 + ((lane >> 4) << 3);
  const v4f* rp = (const v4f*)row;          // row is 512B-aligned
  v4f x0 = rp[(kb >> 2) + 0];
  v4f x1 = rp[(kb >> 2) + 1];
  v4f x2 = rp[(kb >> 2) + 4];               // kb + 16
  v4f x3 = rp[(kb >> 2) + 5];
  v16h a;
#pragma unroll
  for (int i = 0; i < 4; ++i) {
    a[i]      = (_Float16)(x0[i] * scale);
    a[4 + i]  = (_Float16)(x1[i] * scale);
    a[8 + i]  = (_Float16)(x2[i] * scale);
    a[12 + i] = (_Float16)(x3[i] * scale);
  }
  return a;
}

// ---------------------------------------------------------------------------
// B-fragment: 32x16 f16 tile from LDS-staged weight W[N][128] (row = out-feat).
// B[k][n] = W[n][k]. Lane l: N = l&15, elems 0..7 = K kb..kb+7, elems 8..15 =
// K kb+16..kb+23 where kb = kt*32 + (l>=16 ? 8 : 0). Two ds_load_b128.
// ---------------------------------------------------------------------------
__device__ inline v16h load_b_frag(const _Float16* sW, int nt, int kt, int lane) {
  int n  = (nt << 4) + (lane & 15);
  int kb = kt * 32 + ((lane >> 4) << 3);
  const v8h* p = (const v8h*)(sW + n * KDIM + kb);   // 16B-aligned
  v8h lo = p[0];
  v8h hi = p[2];
  v16h b;
#pragma unroll
  for (int i = 0; i < 8; ++i) { b[i] = lo[i]; b[8 + i] = hi[i]; }
  return b;
}

// ---------------------------------------------------------------------------
// Y = relu(X @ W^T + bias) ; X:[M,128] f32, W:[N,128] f32, Y:[M,N] f32.
// 256 threads = 8 waves, each wave owns 16 rows x N cols. NT = N/16.
// ---------------------------------------------------------------------------
template <int NT>
__global__ __launch_bounds__(256) void lin_relu_kernel(
    const float* __restrict__ X, const float* __restrict__ W,
    const float* __restrict__ bias, float* __restrict__ Y, int M) {
  constexpr int N = NT * 16;
  __shared__ _Float16 sW[N * KDIM];

  int tid = threadIdx.x;
#pragma unroll 4
  for (int i = tid; i < N * KDIM; i += 256) sW[i] = (_Float16)W[i];
  __syncthreads();

  int wave = tid >> 5, lane = tid & 31;
  int m0 = blockIdx.x * 128 + wave * 16;
  if (m0 >= M) return;                       // wave-uniform: EXEC stays all-1s

  int m = m0 + (lane & 15);
  if (m >= M) m = M - 1;                     // clamp reads; stores are guarded
  const float* row = X + (size_t)m * KDIM;

  v8f acc[NT];
#pragma unroll
  for (int nt = 0; nt < NT; ++nt) acc[nt] = (v8f){0,0,0,0,0,0,0,0};

#pragma unroll
  for (int kt = 0; kt < KDIM / 32; ++kt) {
    v16h a = load_a_frag(row, kt, lane, 1.0f);
#pragma unroll
    for (int nt = 0; nt < NT; ++nt) {
      v16h b = load_b_frag(sW, nt, kt, lane);
      acc[nt] = __builtin_amdgcn_wmma_f32_16x16x32_f16(
          false, a, false, b, (short)0, acc[nt], false, false);
    }
  }

  int mrow = m0 + ((lane >> 4) << 3);        // C/D layout: lanes 16-31 -> M+8
#pragma unroll
  for (int nt = 0; nt < NT; ++nt) {
    int n = (nt << 4) + (lane & 15);
    float bv = bias[n];
#pragma unroll
    for (int v = 0; v < 8; ++v) {
      int mm = mrow + v;
      if (mm < M) {
        float r = acc[nt][v] + bv;
        Y[(size_t)mm * N + n] = r > 0.0f ? r : 0.0f;
      }
    }
  }
}

// ---------------------------------------------------------------------------
// Y = (Agg * 1/max(deg,1)) @ Wl^T + bias + Xr @ Wr^T   (no activation)
// ---------------------------------------------------------------------------
template <int NT>
__global__ __launch_bounds__(256) void sage_out_kernel(
    const float* __restrict__ Agg, const float* __restrict__ Deg,
    const float* __restrict__ Xr,
    const float* __restrict__ Wl, const float* __restrict__ Wr,
    const float* __restrict__ bias, float* __restrict__ Y, int M) {
  constexpr int N = NT * 16;
  __shared__ _Float16 sWl[N * KDIM];
  __shared__ _Float16 sWr[N * KDIM];

  int tid = threadIdx.x;
#pragma unroll 4
  for (int i = tid; i < N * KDIM; i += 256) {
    sWl[i] = (_Float16)Wl[i];
    sWr[i] = (_Float16)Wr[i];
  }
  __syncthreads();

  int wave = tid >> 5, lane = tid & 31;
  int m0 = blockIdx.x * 128 + wave * 16;
  if (m0 >= M) return;

  int m = m0 + (lane & 15);
  if (m >= M) m = M - 1;
  float scale = 1.0f / fmaxf(Deg[m], 1.0f);  // fused scatter-mean divide
  const float* rowA = Agg + (size_t)m * KDIM;
  const float* rowX = Xr  + (size_t)m * KDIM;

  v8f acc[NT];
#pragma unroll
  for (int nt = 0; nt < NT; ++nt) acc[nt] = (v8f){0,0,0,0,0,0,0,0};

#pragma unroll
  for (int kt = 0; kt < KDIM / 32; ++kt) {   // mean @ Wl^T
    v16h a = load_a_frag(rowA, kt, lane, scale);
#pragma unroll
    for (int nt = 0; nt < NT; ++nt) {
      v16h b = load_b_frag(sWl, nt, kt, lane);
      acc[nt] = __builtin_amdgcn_wmma_f32_16x16x32_f16(
          false, a, false, b, (short)0, acc[nt], false, false);
    }
  }
#pragma unroll
  for (int kt = 0; kt < KDIM / 32; ++kt) {   // + x @ Wr^T
    v16h a = load_a_frag(rowX, kt, lane, 1.0f);
#pragma unroll
    for (int nt = 0; nt < NT; ++nt) {
      v16h b = load_b_frag(sWr, nt, kt, lane);
      acc[nt] = __builtin_amdgcn_wmma_f32_16x16x32_f16(
          false, a, false, b, (short)0, acc[nt], false, false);
    }
  }

  int mrow = m0 + ((lane >> 4) << 3);
#pragma unroll
  for (int nt = 0; nt < NT; ++nt) {
    int n = (nt << 4) + (lane & 15);
    float bv = bias[n];
#pragma unroll
    for (int v = 0; v < 8; ++v) {
      int mm = mrow + v;
      if (mm < M) Y[(size_t)mm * N + n] = acc[nt][v] + bv;
    }
  }
}

// ---------------------------------------------------------------------------
// Edge scatter: agg[dst] += w_e * x[src]  (+ optional degree count).
// One wave per edge; each lane moves 4 features (16B load + 4 f32 atomics,
// resolved in the 192MB L2 where agg resides).
// ---------------------------------------------------------------------------
__global__ __launch_bounds__(256) void scatter_kernel(
    const float* __restrict__ X, const long long* __restrict__ ei,
    const float* __restrict__ ew, float* __restrict__ agg,
    float* __restrict__ deg, int E) {
  int e = blockIdx.x * 8 + (threadIdx.x >> 5);
  if (e >= E) return;                        // wave-uniform
  int lane = threadIdx.x & 31;

  long long s = ei[e];
  long long d = ei[(size_t)E + e];
  float w = ew[e];

  const v4f* xs = (const v4f*)(X + (size_t)s * KDIM);
  v4f v = xs[lane];

  float* out = agg + (size_t)d * KDIM + lane * 4;
  atomicAdd(out + 0, v[0] * w);
  atomicAdd(out + 1, v[1] * w);
  atomicAdd(out + 2, v[2] * w);
  atomicAdd(out + 3, v[3] * w);

  if (deg != nullptr && lane == 0) atomicAdd(deg + d, 1.0f);
}

// ---------------------------------------------------------------------------
extern "C" void kernel_launch(void* const* d_in, const int* in_sizes, int n_in,
                              void* d_out, int out_size, void* d_ws, size_t ws_size,
                              hipStream_t stream) {
  const float*     x      = (const float*)d_in[0];
  const long long* ei     = (const long long*)d_in[1];   // int64 [2,E]
  const float*     ew     = (const float*)d_in[2];
  const float*     ln1_w  = (const float*)d_in[3];
  const float*     ln1_b  = (const float*)d_in[4];
  const float*     c1_l_w = (const float*)d_in[5];
  const float*     c1_l_b = (const float*)d_in[6];
  const float*     c1_r_w = (const float*)d_in[7];
  const float*     ln2_w  = (const float*)d_in[8];
  const float*     ln2_b  = (const float*)d_in[9];
  const float*     c2_l_w = (const float*)d_in[10];
  const float*     c2_l_b = (const float*)d_in[11];
  const float*     c2_r_w = (const float*)d_in[12];

  const int M = in_sizes[0] / KDIM;          // 100000 nodes
  const int E = in_sizes[2];                 // 600000 edges

  char* ws = (char*)d_ws;
  size_t NF = (size_t)M * KDIM * sizeof(float);
  float* bufA = (float*)(ws);                // lx1 / lx2
  float* bufB = (float*)(ws + NF);           // agg1 / agg2
  float* bufC = (float*)(ws + 2 * NF);       // rst1
  float* deg  = (float*)(ws + 3 * NF);

  const int gM = (M + 127) / 128;
  const int gE = (E + 7) / 8;

  // layer 1: lx1 = relu(x @ ln1_w^T + ln1_b)
  hipMemsetAsync(bufB, 0, NF, stream);
  hipMemsetAsync(deg, 0, (size_t)M * sizeof(float), stream);
  lin_relu_kernel<8><<<gM, 256, 0, stream>>>(x, ln1_w, ln1_b, bufA, M);

  // conv1: scatter-mean + dual GEMM -> rst1
  scatter_kernel<<<gE, 256, 0, stream>>>(bufA, ei, ew, bufB, deg, E);
  sage_out_kernel<8><<<gM, 256, 0, stream>>>(bufB, deg, bufA,
                                             c1_l_w, c1_r_w, c1_l_b, bufC, M);

  // layer 2: lx2 = relu(rst1 @ ln2_w^T + ln2_b)
  lin_relu_kernel<8><<<gM, 256, 0, stream>>>(bufC, ln2_w, ln2_b, bufA, M);

  // conv2: scatter-mean + dual GEMM -> out [M,64]
  hipMemsetAsync(bufB, 0, NF, stream);
  scatter_kernel<<<gE, 256, 0, stream>>>(bufA, ei, ew, bufB, nullptr, E);
  sage_out_kernel<4><<<gM, 256, 0, stream>>>(bufB, deg, bufA,
                                             c2_l_w, c2_r_w, c2_l_b,
                                             (float*)d_out, M);
}
